// MoEModel_44487271252125
// MI455X (gfx1250) — compile-verified, matrix operand order checked by reading, not run on hardware
//
#include <hip/hip_runtime.h>
#include <hip/hip_bf16.h>
#include <math.h>

// ---------------------------------------------------------------------------
// Sizes (fixed by the reference)
// ---------------------------------------------------------------------------
#define BATCH   128
#define DDIM    12544          // 14*14*64
#define EDIM    8
#define CDIM    1000
#define CTILES  63             // ceil(1000/16)

typedef __bf16 bf16_t;
typedef bf16_t v16bf __attribute__((ext_vector_type(16)));
typedef float  v8f   __attribute__((ext_vector_type(8)));
typedef unsigned int u32x4 __attribute__((ext_vector_type(4)));
typedef int          i32x4 __attribute__((ext_vector_type(4)));
typedef int          i32x8 __attribute__((ext_vector_type(8)));

union ABits { uint4 q[2]; v16bf v; };

#if defined(__AMDGCN__) && __has_builtin(__builtin_amdgcn_tensor_load_to_lds)
#define USE_TDM 1
#else
#define USE_TDM 0
#endif

__device__ __forceinline__ unsigned short f2bf(float f) {
    unsigned int u = __float_as_uint(f);
    unsigned int r = u + 0x7FFFu + ((u >> 16) & 1u);   // round-to-nearest-even
    return (unsigned short)(r >> 16);
}

#if USE_TDM
// Issue a TDM DMA of a 32x16 fp32 tile (rows stride = CDIM floats) into LDS.
// rem_c clamps tensor_dim0 so hardware zero-fills columns past the tensor edge.
__device__ __forceinline__ void tdm_issue(const float* tile, unsigned lds_off, int rem_c)
{
    unsigned long long ga = (unsigned long long)(uintptr_t)tile;
    u32x4 g0;
    g0[0] = 1u;                                             // count=1, user mode
    g0[1] = lds_off;                                        // LDS byte address
    g0[2] = (unsigned)(ga & 0xFFFFFFFFull);                 // global_addr[31:0]
    g0[3] = (unsigned)((ga >> 32) & 0x01FFFFFFull)          // global_addr[56:32]
          | 0x80000000u;                                    // type = 2 ("image")
    i32x8 g1;
    g1[0] = 0x20000;                                        // data_size = 4 bytes
    g1[1] = (rem_c & 0xFFFF) << 16;                         // tensor_dim0[15:0]
    g1[2] = (rem_c >> 16) | (DDIM << 16);                   // dim0 hi | tensor_dim1 lo
    g1[3] = (DDIM >> 16) | (16 << 16);                      // dim1 hi | tile_dim0 = 16
    g1[4] = 32;                                             // tile_dim1 = 32, tile_dim2 = 0
    g1[5] = CDIM;                                           // tensor_dim0_stride = 1000
    g1[6] = 0;
    g1[7] = 0;
    i32x4 z4 = {0, 0, 0, 0};
#if defined(__clang_major__) && (__clang_major__ >= 23)
    i32x8 z8 = {0, 0, 0, 0, 0, 0, 0, 0};
    __builtin_amdgcn_tensor_load_to_lds(g0, g1, z4, z4, z8, 0);
#else
    __builtin_amdgcn_tensor_load_to_lds(g0, g1, z4, z4, 0);
#endif
}
#endif

// ---------------------------------------------------------------------------
// K1: conv1 (3x3, 1->32, VALID) + relu + maxpool2  ->  h1 [128,31,31,32]
// ---------------------------------------------------------------------------
__global__ void conv1_kernel(const float* __restrict__ x,
                             const float* __restrict__ w1,
                             const float* __restrict__ b1,
                             float* __restrict__ h1)
{
    int tid = blockIdx.x * blockDim.x + threadIdx.x;
    const int total = BATCH * 31 * 31 * 32;
    if (tid >= total) return;
    int c  = tid & 31;
    int t  = tid >> 5;
    int xo = t % 31;  t /= 31;
    int yo = t % 31;  t /= 31;
    int b  = t;

    float m = -3.4e38f;
    #pragma unroll
    for (int dy = 0; dy < 2; ++dy)
    #pragma unroll
    for (int dx = 0; dx < 2; ++dx) {
        int py = 2 * yo + dy, px = 2 * xo + dx;     // conv-output coords 0..61
        float acc = b1[c];
        #pragma unroll
        for (int ky = 0; ky < 3; ++ky)
        #pragma unroll
        for (int kx = 0; kx < 3; ++kx)
            acc = fmaf(x[(b * 64 + py + ky) * 64 + (px + kx)],
                       w1[(ky * 3 + kx) * 32 + c], acc);
        acc = fmaxf(acc, 0.f);
        m = fmaxf(m, acc);
    }
    h1[tid] = m;   // [b][y][x][c], c fastest
}

// ---------------------------------------------------------------------------
// K2: conv2 (3x3, 32->64, VALID) + relu + maxpool2 -> h (f32) and hbf (bf16)
// ---------------------------------------------------------------------------
__global__ void conv2_kernel(const float* __restrict__ h1,
                             const float* __restrict__ w2,
                             const float* __restrict__ b2,
                             float* __restrict__ h,
                             unsigned short* __restrict__ hbf)
{
    int c   = threadIdx.x;                 // 0..63
    int blk = blockIdx.x;
    int xo  = blk % 14;  blk /= 14;
    int yo  = blk % 14;  blk /= 14;
    int b   = blk;

    float m = -3.4e38f;
    #pragma unroll
    for (int dy = 0; dy < 2; ++dy)
    #pragma unroll
    for (int dx = 0; dx < 2; ++dx) {
        int py = 2 * yo + dy, px = 2 * xo + dx;     // 0..27 (+2 stays < 31)
        float acc = b2[c];
        for (int ky = 0; ky < 3; ++ky)
        for (int kx = 0; kx < 3; ++kx) {
            const float* hp = &h1[((b * 31 + py + ky) * 31 + (px + kx)) * 32];
            const float* wp = &w2[((ky * 3 + kx) * 32) * 64 + c];
            #pragma unroll
            for (int ci = 0; ci < 32; ++ci)
                acc = fmaf(hp[ci], wp[ci * 64], acc);
        }
        acc = fmaxf(acc, 0.f);
        m = fmaxf(m, acc);
    }
    int d   = (yo * 14 + xo) * 64 + c;     // matches reshape(B,-1) of NHWC
    int idx = b * DDIM + d;
    h[idx]   = m;
    hbf[idx] = f2bf(m);
}

// ---------------------------------------------------------------------------
// K3: gating -> dense gate-weight matrix Wmat[b][e] (0 outside top-2)
// ---------------------------------------------------------------------------
__global__ void gate_kernel(const float* __restrict__ h,
                            const float* __restrict__ gw,
                            const float* __restrict__ gb,
                            float* __restrict__ Wmat)
{
    __shared__ float red[256][EDIM];
    int b = blockIdx.x, t = threadIdx.x;

    float acc[EDIM];
    #pragma unroll
    for (int e = 0; e < EDIM; ++e) acc[e] = 0.f;

    const float* hb = h + b * DDIM;
    for (int d = t; d < DDIM; d += 256) {
        float hv = hb[d];
        const float* g = gw + d * EDIM;
        #pragma unroll
        for (int e = 0; e < EDIM; ++e) acc[e] = fmaf(hv, g[e], acc[e]);
    }
    #pragma unroll
    for (int e = 0; e < EDIM; ++e) red[t][e] = acc[e];
    __syncthreads();
    for (int s = 128; s > 0; s >>= 1) {
        if (t < s) {
            #pragma unroll
            for (int e = 0; e < EDIM; ++e) red[t][e] += red[t + s][e];
        }
        __syncthreads();
    }
    if (t == 0) {
        float lg[EDIM];
        float mx = -3.4e38f;
        #pragma unroll
        for (int e = 0; e < EDIM; ++e) { lg[e] = red[0][e] + gb[e]; mx = fmaxf(mx, lg[e]); }
        float sum = 0.f;
        #pragma unroll
        for (int e = 0; e < EDIM; ++e) { lg[e] = __expf(lg[e] - mx); sum += lg[e]; }
        float inv = 1.f / sum;
        #pragma unroll
        for (int e = 0; e < EDIM; ++e) lg[e] *= inv;
        int i0 = 0;
        for (int e = 1; e < EDIM; ++e) if (lg[e] > lg[i0]) i0 = e;
        int i1 = (i0 == 0) ? 1 : 0;
        for (int e = 0; e < EDIM; ++e) if (e != i0 && lg[e] > lg[i1]) i1 = e;
        #pragma unroll
        for (int e = 0; e < EDIM; ++e)
            Wmat[b * EDIM + e] = (e == i0 || e == i1) ? lg[e] : 0.f;
    }
}

// ---------------------------------------------------------------------------
// K4: zero the combine accumulator
// ---------------------------------------------------------------------------
__global__ void zero_kernel(float* __restrict__ p, int n)
{
    int i = blockIdx.x * blockDim.x + threadIdx.x;
    if (i < n) p[i] = 0.f;
}

// ---------------------------------------------------------------------------
// K5: expert GEMM with bf16 WMMA + TDM async weight streaming.
//   grid  = (63 c-tiles, 8 experts), block = 256 (8 waves)
//   wave w handles rows 16w..16w+15; all waves share a 32x16 B tile.
//   Pipeline: TDM DMA fp32 tile (double buffered) -> bf16 re-layout in LDS
//             -> ds_load_b128 fragments -> v_wmma_f32_16x16x32_bf16.
// ---------------------------------------------------------------------------
__global__ void __launch_bounds__(256)
expert_kernel(const unsigned short* __restrict__ hbf,
              const float* __restrict__ ew,
              const float* __restrict__ eb,
              const float* __restrict__ Wmat,
              float* __restrict__ comb)
{
    __shared__ __align__(16) float          ldsF[2][32 * 16];   // raw fp32 tiles (TDM dest)
    __shared__ __align__(16) unsigned short ldsB[2][32 * 16];   // bf16 WMMA B layout

    const int ct   = blockIdx.x;            // c tile
    const int e    = blockIdx.y;            // expert
    const int c0   = ct * 16;
    const int t    = threadIdx.x;
    const int wv   = t >> 5;                // wave -> b tile
    const int lane = t & 31;
    const int g    = lane >> 4;             // lane half-group
    const int mrow = lane & 15;             // A row within tile
    const int brow = wv * 16 + mrow;

    const unsigned short* aPtr0 = hbf + (size_t)brow * DDIM + g * 8;
    const unsigned short* aPtr1 = aPtr0 + 16;
    const float* wbase = ew + (size_t)e * DDIM * CDIM;
    const int rem_c = CDIM - c0;            // hardware zero-fills columns >= rem_c

    // cooperative convert decomposition: element idx -> (k = idx>>4, n = idx&15)
    const int k_a = t >> 4,         n_a = t & 15;       // idx = t
    const int k_b = (t + 256) >> 4, n_b = t & 15;       // idx = t + 256
    const int hiA = k_a >> 4, ggA = (k_a >> 3) & 1, jA = hiA * 8 + (k_a & 7);
    const int hiB = k_b >> 4, ggB = (k_b >> 3) & 1, jB = hiB * 8 + (k_b & 7);
    const int dstA = (ggA * 16 + n_a) * 16 + jA;
    const int dstB = (ggB * 16 + n_b) * 16 + jB;

    v8f acc = {};

#if USE_TDM
    const unsigned ldsF0 = (unsigned)(uintptr_t)(void*)&ldsF[0][0];
    const unsigned ldsF1 = (unsigned)(uintptr_t)(void*)&ldsF[1][0];

    if (wv == 0)                                            // one DMA per workgroup
        tdm_issue(wbase + c0, ldsF0, rem_c);                // tile for k0 = 0

    for (int k0 = 0; k0 < DDIM; k0 += 32) {
        const int buf = (k0 >> 5) & 1;
        if (wv == 0) {
            if (k0 + 32 < DDIM) {                           // prefetch next tile
                tdm_issue(wbase + (size_t)(k0 + 32) * CDIM + c0,
                          buf ? ldsF0 : ldsF1, rem_c);
                __builtin_amdgcn_s_wait_tensorcnt(1);       // current tile landed
            } else {
                __builtin_amdgcn_s_wait_tensorcnt(0);
            }
        }
        __syncthreads();                                    // publish fp32 tile

        // fp32 -> bf16, scatter into WMMA B-operand lane layout
        ldsB[buf][dstA] = f2bf(ldsF[buf][t]);
        ldsB[buf][dstB] = f2bf(ldsF[buf][t + 256]);
        __syncthreads();

        ABits A, B;
        A.q[0] = *(const uint4*)(aPtr0 + k0);
        A.q[1] = *(const uint4*)(aPtr1 + k0);
        const uint4* bp = (const uint4*)&ldsB[buf][lane * 16];
        B.q[0] = bp[0];
        B.q[1] = bp[1];
        acc = __builtin_amdgcn_wmma_f32_16x16x32_bf16(
                  false, A.v, false, B.v, (short)0, acc, false, false);
    }
#else
    // Fallback: stage weights through per-lane global loads (clamped, unpredicated)
    const int cc_cl = (c0 + n_a < CDIM) ? (c0 + n_a) : (CDIM - 1);
    const float vmul = (c0 + n_a < CDIM) ? 1.f : 0.f;
    for (int k0 = 0; k0 < DDIM; k0 += 32) {
        float wa = wbase[(size_t)(k0 + k_a) * CDIM + cc_cl] * vmul;
        float wb = wbase[(size_t)(k0 + k_b) * CDIM + cc_cl] * vmul;
        ldsB[0][dstA] = f2bf(wa);
        ldsB[0][dstB] = f2bf(wb);
        __syncthreads();
        ABits A, B;
        A.q[0] = *(const uint4*)(aPtr0 + k0);
        A.q[1] = *(const uint4*)(aPtr1 + k0);
        const uint4* bp = (const uint4*)&ldsB[0][lane * 16];
        B.q[0] = bp[0];
        B.q[1] = bp[1];
        acc = __builtin_amdgcn_wmma_f32_16x16x32_bf16(
                  false, A.v, false, B.v, (short)0, acc, false, false);
        __syncthreads();
    }
#endif

    // ---- epilogue: gate-weight * (acc + bias) -> atomic combine
    const int n  = lane & 15;
    const int cc = c0 + n;
    if (cc < CDIM) {
        float bias = eb[e * CDIM + cc];
        #pragma unroll
        for (int r = 0; r < 8; ++r) {
            int m2 = r + 8 * g;                 // C/D layout: lanes16-31 hold M=8..15
            int b2 = wv * 16 + m2;
            float gwv = Wmat[b2 * EDIM + e];
            if (gwv != 0.f)
                atomicAdd(&comb[(size_t)b2 * CDIM + cc], gwv * (acc[r] + bias));
        }
    }
}

// ---------------------------------------------------------------------------
// K6: row softmax over C=1000
// ---------------------------------------------------------------------------
__global__ void softmax_kernel(const float* __restrict__ comb,
                               float* __restrict__ out)
{
    __shared__ float red[256];
    int b = blockIdx.x, t = threadIdx.x;
    const float* row = comb + (size_t)b * CDIM;

    float mx = -3.4e38f;
    for (int c = t; c < CDIM; c += 256) mx = fmaxf(mx, row[c]);
    red[t] = mx; __syncthreads();
    for (int s = 128; s > 0; s >>= 1) {
        if (t < s) red[t] = fmaxf(red[t], red[t + s]);
        __syncthreads();
    }
    mx = red[0];
    __syncthreads();

    float sum = 0.f;
    for (int c = t; c < CDIM; c += 256) sum += __expf(row[c] - mx);
    red[t] = sum; __syncthreads();
    for (int s = 128; s > 0; s >>= 1) {
        if (t < s) red[t] += red[t + s];
        __syncthreads();
    }
    sum = red[0];
    __syncthreads();

    float inv = 1.f / sum;
    for (int c = t; c < CDIM; c += 256)
        out[(size_t)b * CDIM + c] = __expf(row[c] - mx) * inv;
}

// ---------------------------------------------------------------------------
// launch
// ---------------------------------------------------------------------------
static inline size_t align256(size_t x) { return (x + 255) & ~(size_t)255; }

extern "C" void kernel_launch(void* const* d_in, const int* in_sizes, int n_in,
                              void* d_out, int out_size, void* d_ws, size_t ws_size,
                              hipStream_t stream)
{
    const float* x   = (const float*)d_in[0];
    const float* c1w = (const float*)d_in[1];
    const float* c1b = (const float*)d_in[2];
    const float* c2w = (const float*)d_in[3];
    const float* c2b = (const float*)d_in[4];
    const float* gw  = (const float*)d_in[5];
    const float* gb  = (const float*)d_in[6];
    const float* ewp = (const float*)d_in[7];
    const float* ebp = (const float*)d_in[8];
    float* out = (float*)d_out;

    char* ws = (char*)d_ws;
    size_t off = 0;
    float* h1 = (float*)(ws + off);                  off = align256(off + (size_t)BATCH*31*31*32*4);
    float* h  = (float*)(ws + off);                  off = align256(off + (size_t)BATCH*DDIM*4);
    unsigned short* hbf = (unsigned short*)(ws+off); off = align256(off + (size_t)BATCH*DDIM*2);
    float* Wm = (float*)(ws + off);                  off = align256(off + (size_t)BATCH*EDIM*4);
    float* comb = (float*)(ws + off);                off = align256(off + (size_t)BATCH*CDIM*4);
    (void)ws_size; (void)n_in; (void)in_sizes; (void)out_size;

    {   // conv1
        int total = BATCH * 31 * 31 * 32;
        conv1_kernel<<<(total + 255) / 256, 256, 0, stream>>>(x, c1w, c1b, h1);
    }
    {   // conv2 + flatten + bf16 copy
        conv2_kernel<<<BATCH * 14 * 14, 64, 0, stream>>>(h1, c2w, c2b, h, hbf);
    }
    {   // gate -> dense top-2 weight matrix
        gate_kernel<<<BATCH, 256, 0, stream>>>(h, gw, gb, Wm);
    }
    {   // zero combine accumulator (fresh every call)
        int n = BATCH * CDIM;
        zero_kernel<<<(n + 255) / 256, 256, 0, stream>>>(comb, n);
    }
    {   // expert GEMM (WMMA bf16 + TDM weight streaming)
        dim3 grid(CTILES, EDIM);
        expert_kernel<<<grid, 256, 0, stream>>>(hbf, ewp, ebp, Wm, comb);
    }
    {   // final softmax
        softmax_kernel<<<BATCH, 256, 0, stream>>>(comb, out);
    }
}
